// SCAttention_25975962206470
// MI455X (gfx1250) — matrix operand order repaired and checked by reading, not compiled, not run on hardware
//
#include <hip/hip_runtime.h>

// Problem sizes from the reference
#define Bsz 8
#define Psz 2048
#define Qsz 512
#define Hsz 1024
#define WAVES 4           // waves per block; each wave owns 16 P-rows

// f16 LDS row strides, padded so row byte-stride = 4 dwords mod 64 banks
#define SA 1032           // passage / attn per-wave tile  (16 x 1024 cols)
#define SS 520            // scores / alpha per-wave tile  (16 x 512 cols)
#define SC 1032           // shared stage: 32 rows x 1024 cols (question/W rows)
#define ST 520            // shared stage: 64 rows x 512 cols (question^T)

typedef __attribute__((ext_vector_type(16))) _Float16 v16h;
typedef __attribute__((ext_vector_type(4)))  _Float16 v4h;
typedef __attribute__((ext_vector_type(8)))  float    v8f;

static __device__ __forceinline__ v8f wmma_f16(v16h a, v16h b, v8f c) {
    return __builtin_amdgcn_wmma_f32_16x16x32_f16(false, a, false, b,
                                                  (short)0, c, false, false);
}

// ---- fragment loaders (layouts per CDNA5 ISA 7.12.2) -----------------------

// A-matrix 16x32 frag from an LDS f16 tile (tile = &M[0][0], ld = f16 stride).
// lanes 0-15: M=lane, K=kb+{0..7,16..23}; lanes 16-31: M=lane-16, K=kb+{8..15,24..31}
static __device__ __forceinline__ v16h a_frag_lds(const _Float16* tile, int ld,
                                                  int kb, int lane) {
    const _Float16* p = tile + (lane & 15) * ld + kb + ((lane & 16) ? 8 : 0);
    v16h a;
#pragma unroll
    for (int i = 0; i < 8; ++i) a[i] = p[i];
#pragma unroll
    for (int i = 0; i < 8; ++i) a[8 + i] = p[16 + i];
    return a;
}

// B-matrix 32x16 frag where B[k][n] = src[n][k], src is an LDS f16 tile.
// lanes 0-15: N=lane, K=kb+{0..7,16..23}; lanes 16-31: N=lane-16, K=kb+{8..15,24..31}
static __device__ __forceinline__ v16h b_frag_lds(const _Float16* tile, int ld,
                                                  int kb, int lane) {
    const _Float16* p = tile + (lane & 15) * ld + kb + ((lane & 16) ? 8 : 0);
    v16h b;
#pragma unroll
    for (int i = 0; i < 8; ++i) b[i] = p[i];
#pragma unroll
    for (int i = 0; i < 8; ++i) b[8 + i] = p[16 + i];
    return b;
}

// Cooperative stage: 32 rows x 1024 f32 -> LDS f16 chunk (128 threads: 4/row).
static __device__ __forceinline__ void stage32(_Float16* dst, const float* src,
                                               int tid) {
    const int row = tid >> 2;            // 0..31
    const int c0  = (tid & 3) * 256;     // 4 threads cover 1024 cols
    const float4* s = (const float4*)(src + (size_t)row * Hsz + c0);
    v4h* d = (v4h*)(dst + row * SC + c0);
#pragma unroll 8
    for (int i = 0; i < 64; ++i) {
        float4 f = s[i];
        v4h h;
        h[0] = (_Float16)f.x; h[1] = (_Float16)f.y;
        h[2] = (_Float16)f.z; h[3] = (_Float16)f.w;
        d[i] = h;
    }
}

// Cooperative transpose-stage: dst[h][q] = src[q][h0+h] for h in [0,64), q in [0,512).
// Thread t reads 4 consecutive q-rows coalesced, writes v4h (4 q's) per h:
// one conflict-free ds_store_b64 per (thread, h).
static __device__ __forceinline__ void stage_qT(_Float16* dst, const float* src,
                                                int h0, int tid) {
    const int q0 = tid * 4;              // 128 threads x 4 rows = 512 q
    const float* s0 = src + (size_t)(q0 + 0) * Hsz + h0;
    const float* s1 = src + (size_t)(q0 + 1) * Hsz + h0;
    const float* s2 = src + (size_t)(q0 + 2) * Hsz + h0;
    const float* s3 = src + (size_t)(q0 + 3) * Hsz + h0;
#pragma unroll 4
    for (int h = 0; h < 64; h += 4) {
        float4 f0 = *(const float4*)(s0 + h);
        float4 f1 = *(const float4*)(s1 + h);
        float4 f2 = *(const float4*)(s2 + h);
        float4 f3 = *(const float4*)(s3 + h);
        v4h a, b, c, d;
        a[0] = (_Float16)f0.x; a[1] = (_Float16)f1.x; a[2] = (_Float16)f2.x; a[3] = (_Float16)f3.x;
        b[0] = (_Float16)f0.y; b[1] = (_Float16)f1.y; b[2] = (_Float16)f2.y; b[3] = (_Float16)f3.y;
        c[0] = (_Float16)f0.z; c[1] = (_Float16)f1.z; c[2] = (_Float16)f2.z; c[3] = (_Float16)f3.z;
        d[0] = (_Float16)f0.w; d[1] = (_Float16)f1.w; d[2] = (_Float16)f2.w; d[3] = (_Float16)f3.w;
        *(v4h*)(dst + (h + 0) * ST + q0) = a;
        *(v4h*)(dst + (h + 1) * ST + q0) = b;
        *(v4h*)(dst + (h + 2) * ST + q0) = c;
        *(v4h*)(dst + (h + 3) * ST + q0) = d;
    }
}

// ---- fused attention + linear kernel ---------------------------------------

__global__ __launch_bounds__(WAVES * 32) void scatt_fused_kernel(
    const float* __restrict__ passage,   // [B,P,H]
    const float* __restrict__ question,  // [B,Q,H]
    const int*   __restrict__ qmask,     // [B,Q]
    const float* __restrict__ mapW,      // [H,H]   y = x @ W^T + b
    const float* __restrict__ mapb,      // [H]
    float*       __restrict__ out)       // [B,P,H]
{
    extern __shared__ char smem[];
    // per-wave passage f16 (phase 1), overlaid by attn f16 (phases 3-4)
    _Float16* ldsA = (_Float16*)smem;                                  // 4*16*SA*2 = 132096 B
    // per-wave scores f16 (phase 1-2), alpha written in place (phases 2-3)
    _Float16* ldsS = (_Float16*)(smem + WAVES * 16 * SA * 2);          // 4*16*SS*2 =  66560 B
    // shared staging buffer: 32xSC rows (phases 1,4) or 64xST transposed (phase 3)
    _Float16* chunk = (_Float16*)(smem + WAVES * 16 * (SA + SS) * 2);  // 64*ST*2   =  66560 B

    const int tid  = threadIdx.x;
    const int lane = tid & 31;
    const int w    = tid >> 5;
    const int b    = blockIdx.y;
    const int p0   = blockIdx.x * (WAVES * 16) + w * 16;

    _Float16* Apane = ldsA + w * 16 * SA;   // passage f16, later attn f16
    _Float16* Spane = ldsS + w * 16 * SS;   // scores f16, later alpha f16

    const float* Ptile = passage + ((size_t)b * Psz + p0) * Hsz;
    const float* Qbase = question + (size_t)b * Qsz * Hsz;

    const int n  = lane & 15;
    const int mb = (lane >> 4) * 8;         // C/D tile: m = mb + r, n = lane&15

    // ---- Phase 0: convert this wave's 16x1024 passage tile to f16 in LDS (once)
    {
        const int row = lane >> 1;
        const int c0  = (lane & 1) * 512;
        const float4* s = (const float4*)(Ptile + (size_t)row * Hsz + c0);
        v4h* d = (v4h*)(Apane + row * SA + c0);
#pragma unroll 8
        for (int i = 0; i < 128; ++i) {
            float4 f = s[i];
            v4h h;
            h[0] = (_Float16)f.x; h[1] = (_Float16)f.y;
            h[2] = (_Float16)f.z; h[3] = (_Float16)f.w;
            d[i] = h;
        }
    }

    // ---- Phase 1: scores[16x512] = P_tile @ Q^T, question rows staged per block
    for (int qc = 0; qc < Qsz / 32; ++qc) {
        __syncthreads();                                   // chunk reuse safety
        stage32(chunk, Qbase + (size_t)(qc * 32) * Hsz, tid);
        __syncthreads();

        v8f acc0 = {}, acc1 = {};
#pragma unroll 4
        for (int ks = 0; ks < Hsz / 32; ++ks) {
            v16h av = a_frag_lds(Apane, SA, ks * 32, lane);
            acc0 = wmma_f16(av, b_frag_lds(chunk,           SC, ks * 32, lane), acc0);
            acc1 = wmma_f16(av, b_frag_lds(chunk + 16 * SC, SC, ks * 32, lane), acc1);
        }
#pragma unroll
        for (int r = 0; r < 8; ++r) {
            Spane[(mb + r) * SS + (qc * 2 + 0) * 16 + n] = (_Float16)acc0[r];
            Spane[(mb + r) * SS + (qc * 2 + 1) * 16 + n] = (_Float16)acc1[r];
        }
    }

    // ---- Phase 2: masked softmax + renorm; alpha overwrites scores in place
    // alpha_q = m_q*exp(s_q*m_q - max(s*m)) / (sum m*exp(s*m - max) + 1e-13)
    {
        const int row = lane & 15;
        const int q0  = (lane >> 4) * 256;     // two lanes per row
        _Float16* srow = Spane + row * SS;
        const int* mrow = qmask + b * Qsz;

        float mx = -3.0e38f;
        for (int q = q0; q < q0 + 256; ++q) {
            float m = (float)mrow[q];
            mx = fmaxf(mx, (float)srow[q] * m);
        }
        mx = fmaxf(mx, __shfl_xor(mx, 16, 32));

        float sum = 0.0f;
        for (int q = q0; q < q0 + 256; ++q) {
            float m = (float)mrow[q];
            sum += m * __expf((float)srow[q] * m - mx);
        }
        sum += __shfl_xor(sum, 16, 32);
        float inv = 1.0f / (sum + 1e-13f);

        for (int q = q0; q < q0 + 256; ++q) {
            float m = (float)mrow[q];
            srow[q] = (_Float16)(m * __expf((float)srow[q] * m - mx) * inv);
        }
    }

    // ---- Phase 3: attn[16x1024] = alpha @ Q  (overlays dead passage region)
    // question^T staged 64 h-cols at a time; inner loop is pure ds + wmma.
    for (int htg = 0; htg < Hsz / 64; ++htg) {
        __syncthreads();                                   // protect chunk
        stage_qT(chunk, Qbase, htg * 64, tid);
        __syncthreads();

        v8f acc0 = {}, acc1 = {}, acc2 = {}, acc3 = {};
#pragma unroll 2
        for (int ks = 0; ks < Qsz / 32; ++ks) {
            v16h av = a_frag_lds(Spane, SS, ks * 32, lane);
            acc0 = wmma_f16(av, b_frag_lds(chunk,           ST, ks * 32, lane), acc0);
            acc1 = wmma_f16(av, b_frag_lds(chunk + 16 * ST, ST, ks * 32, lane), acc1);
            acc2 = wmma_f16(av, b_frag_lds(chunk + 32 * ST, ST, ks * 32, lane), acc2);
            acc3 = wmma_f16(av, b_frag_lds(chunk + 48 * ST, ST, ks * 32, lane), acc3);
        }
#pragma unroll
        for (int r = 0; r < 8; ++r) {
            _Float16* o = Apane + (mb + r) * SA + htg * 64 + n;
            o[0]  = (_Float16)acc0[r];
            o[16] = (_Float16)acc1[r];
            o[32] = (_Float16)acc2[r];
            o[48] = (_Float16)acc3[r];
        }
    }

    // ---- Phase 4: y = relu(attn @ W^T + bias); W rows staged per block
    for (int og = 0; og < Hsz / 32; ++og) {
        __syncthreads();                                  // protect chunk
        stage32(chunk, mapW + (size_t)(og * 32) * Hsz, tid);
        __builtin_prefetch(mapW + (size_t)(((og + 1) & 31) * 32) * Hsz, 0, 1);
        __syncthreads();

        v8f acc0 = {}, acc1 = {};
#pragma unroll 4
        for (int ks = 0; ks < Hsz / 32; ++ks) {
            v16h av = a_frag_lds(Apane, SA, ks * 32, lane);
            acc0 = wmma_f16(av, b_frag_lds(chunk,           SC, ks * 32, lane), acc0);
            acc1 = wmma_f16(av, b_frag_lds(chunk + 16 * SC, SC, ks * 32, lane), acc1);
        }
        const int   ot0 = og * 2;
        const float b0  = mapb[(ot0 + 0) * 16 + n];
        const float b1  = mapb[(ot0 + 1) * 16 + n];
#pragma unroll
        for (int r = 0; r < 8; ++r) {
            float* o = out + ((size_t)b * Psz + p0 + mb + r) * Hsz + ot0 * 16 + n;
            o[0]  = fmaxf(acc0[r] + b0, 0.0f);
            o[16] = fmaxf(acc1[r] + b1, 0.0f);
        }
    }
}

// ---- host-side launch ------------------------------------------------------

extern "C" void kernel_launch(void* const* d_in, const int* in_sizes, int n_in,
                              void* d_out, int out_size, void* d_ws, size_t ws_size,
                              hipStream_t stream) {
    const float* passage  = (const float*)d_in[0];
    const float* question = (const float*)d_in[1];
    const int*   qmask    = (const int*)d_in[2];
    const float* mapW     = (const float*)d_in[3];
    const float* mapb     = (const float*)d_in[4];
    float*       out      = (float*)d_out;

    dim3 grid(Psz / (WAVES * 16), Bsz);       // (32, 8)
    dim3 block(WAVES * 32);                   // 4 wave32 waves
    size_t lds_bytes = (size_t)WAVES * 16 * SA * 2   // passage/attn f16   132096
                     + (size_t)WAVES * 16 * SS * 2   // scores/alpha f16    66560
                     + (size_t)64 * ST * 2;          // shared stage buf    66560
    // = 265216 B  (< 320 KB/WGP)
    scatt_fused_kernel<<<grid, block, lds_bytes, stream>>>(
        passage, question, qmask, mapW, mapb, out);
}
